// Conv2dfft_27968827031788
// MI455X (gfx1250) — matrix operand order repaired and checked by reading, not compiled, not run on hardware
//
#include <hip/hip_runtime.h>
#include <hip/hip_bf16.h>
#include <cstdint>

// ---- types for WMMA builtins (gfx1250 / CDNA5, wave32) ----
typedef __bf16 bf16_t;
typedef __attribute__((ext_vector_type(16))) __bf16 v16bf;
typedef __attribute__((ext_vector_type(8)))  __bf16 v8bf;
typedef __attribute__((ext_vector_type(8)))  float  v8f;
typedef __attribute__((ext_vector_type(4)))  int    v4i;

#define AS1 __attribute__((address_space(1)))
#define AS3 __attribute__((address_space(3)))

#define C_IN   128
#define F_OUT  128
#define HW     32
#define PHW    34                  // padded spatial (zero halo)
#define K_TOT  (C_IN * 9)          // 1152
#define NT     64                  // pixel tile (2 rows x 32 cols)
#define BS_STRIDE 40               // bf16 per LDS row (80 B; 16B-aligned chunks)
#define BUF_ELEMS (NT * BS_STRIDE) // 2560 bf16 per buffer

#define WPACK_ELEMS (9 * 4 * 128 * 32)             // 147456 bf16
#define WPACK_BYTES (WPACK_ELEMS * 2)              // 294912 B
#define XP_ELEMS    (32 * PHW * PHW * 128)         // padded NHWC bf16
#define XP_BYTES    (XP_ELEMS * 2)                 // 9469952 B

// async global->LDS path (CDNA5)
#if defined(__AMDGCN__) && \
    __has_builtin(__builtin_amdgcn_global_load_async_to_lds_b128) && \
    __has_builtin(__builtin_amdgcn_s_wait_asynccnt)
#define USE_ASYNC_COPY 1
#else
#define USE_ASYNC_COPY 0
#endif

__device__ __forceinline__ unsigned short f2bfbits(float f) {
  unsigned u = __builtin_bit_cast(unsigned, f);
  u += 0x7FFFu + ((u >> 16) & 1u);               // round-to-nearest-even
  return (unsigned short)(u >> 16);
}
__device__ __forceinline__ bf16_t f2bf(float f) {
  unsigned short h = f2bfbits(f);
  return __builtin_bit_cast(bf16_t, h);
}

// ---------------- prep 1: pack weights, fragment-swizzled bf16 ----------------
// wp[((pos*4 + cc)*128 + m)*32 + j];  j 0..15 -> lane-half 0 K runs {0..7,16..23},
//                                     j 16..31 -> lane-half 1 K runs {8..15,24..31}
__global__ __launch_bounds__(256)
void pack_w_kernel(const float* __restrict__ w, bf16_t* __restrict__ wp) {
  const int idx = blockIdx.x * 256 + threadIdx.x;    // 147456 total
  const int j   = idx & 31;
  const int m   = (idx >> 5) & 127;
  const int cc  = (idx >> 12) & 3;
  const int pos = idx >> 14;                         // 0..8
  const int kl  = (j < 8) ? j : (j < 16) ? (j + 8) : (j < 24) ? (j - 8) : j;
  const int c   = cc * 32 + kl;
  wp[idx] = f2bf(w[(size_t)m * K_TOT + c * 9 + pos]);
}

// ---------------- prep 2: x NCHW f32 -> zero-halo NHWC bf16 ----------------
// xp[((n*34 + ph)*34 + pw)*128 + c], interior at ph,pw in [1,32]
__global__ __launch_bounds__(256)
void nhwc_pad_kernel(const float* __restrict__ x, bf16_t* __restrict__ xp) {
  __shared__ __attribute__((aligned(16))) unsigned short T[HW][136];
  const int t  = threadIdx.x;
  const int n  = blockIdx.x / PHW;
  const int ph = blockIdx.x - n * PHW;               // padded row 0..33
  unsigned short* dst = (unsigned short*)xp + (size_t)(n * PHW + ph) * PHW * C_IN;

  if (ph == 0 || ph == PHW - 1) {                    // halo rows: all zero
    for (int c = t; c < PHW * C_IN / 8; c += 256)    // 544 b128 chunks
      ((uint4*)dst)[c] = uint4{0u, 0u, 0u, 0u};
    return;
  }
  const int h  = ph - 1;
  const float* src = x + ((size_t)n * C_IN * HW + h) * HW;
  const int w0 = t & 31;
  const int c0 = t >> 5;
  #pragma unroll
  for (int it = 0; it < 16; ++it) {
    const int c = it * 8 + c0;
    T[w0][c] = f2bfbits(src[(size_t)c * (HW * HW) + w0]);   // coalesced along w
  }
  __syncthreads();
  #pragma unroll
  for (int it = 0; it < 2; ++it) {
    const int chunk = it * 256 + t;                  // 512 chunks of 8 channels
    const int wq = chunk >> 4;
    const int cq = (chunk & 15) << 3;
    *(uint4*)(dst + (wq + 1) * C_IN + cq) = *(const uint4*)&T[wq][cq];
  }
  if (t < 32) {                                      // halo columns 0 and 33
    const int col = (t >> 4) ? (PHW - 1) : 0;
    const int cq  = (t & 15) << 3;
    *(uint4*)(dst + col * C_IN + cq) = uint4{0u, 0u, 0u, 0u};
  }
}

// ---------------- fast main: double-buffered async pipeline, no bounds checks ----------
__global__ __launch_bounds__(256)
void conv_fast(const bf16_t* __restrict__ xp, const bf16_t* __restrict__ wp,
               const float* __restrict__ bias, float* __restrict__ out) {
  __shared__ __attribute__((aligned(16))) bf16_t Bs[2][BUF_ELEMS];   // 2 x 5120 B

  const int t    = threadIdx.x;
  const int wave = t >> 5;
  const int lane = t & 31;
  const int half = lane >> 4;
  const int l16  = lane & 15;

  const int bx    = blockIdx.x;      // 512 = 32 images x 16 row-pair tiles
  const int n_img = bx >> 4;
  const int oh0   = (bx & 15) << 1;

  // B-stage mapping: thread owns (pixel, 8-channel group)
  const int p_st  = t & 63;
  const int cg    = t >> 6;
  const int ow_st = p_st & 31;
  const int oh_st = oh0 + (p_st >> 5);

  // padded-NHWC base for this thread (kh=kw=0 -> padded row oh_st, col ow_st)
  const bf16_t* gb = xp + ((size_t)(n_img * PHW + oh_st) * PHW + ow_st) * C_IN + cg * 8;
  bf16_t* const ld0 = &Bs[0][p_st * BS_STRIDE + cg * 8];
  bf16_t* const ld1 = &Bs[1][p_st * BS_STRIDE + cg * 8];

  // A-fragment base for this lane (32 contiguous bytes per lane per step)
  const int mrow = (wave << 4) + l16;
  const bf16_t* apw = wp + (size_t)mrow * 32 + (half << 4);
  const int koff = half << 3;

  v8f acc[4] = { v8f{}, v8f{}, v8f{}, v8f{} };

  auto stage = [&](int s, bf16_t* dst) {
    const int pos = s >> 2, cc = s & 3;
    const int kh = pos / 3, kw = pos - 3 * kh;
    const bf16_t* gp = gb + ((kh * PHW + kw) * C_IN + cc * 32);
#if USE_ASYNC_COPY
    __builtin_amdgcn_global_load_async_to_lds_b128(
        (AS1 v4i*)(uintptr_t)gp, (AS3 v4i*)(uint32_t)(uintptr_t)dst, 0, 0);
#else
    *(uint4*)dst = *(const uint4*)gp;
#endif
  };

  // prologue: stage chunk 0 into buffer 0
  stage(0, ld0);
#if USE_ASYNC_COPY
  __builtin_amdgcn_s_wait_asynccnt(0);
#endif
  __syncthreads();

  #pragma unroll
  for (int step = 0; step < 36; ++step) {
    // prefetch next chunk into the other buffer (overlaps with WMMAs below)
    if (step < 35) stage(step + 1, ((step + 1) & 1) ? ld1 : ld0);

    // A fragment for this step (L2-resident packed weights)
    const bf16_t* ap = apw + (size_t)step * (128 * 32);
    v8bf a0 = *(const v8bf*)ap;
    v8bf a1 = *(const v8bf*)(ap + 8);
    v16bf afrag;
    #pragma unroll
    for (int i = 0; i < 8; ++i) { afrag[i] = a0[i]; afrag[8 + i] = a1[i]; }

    const bf16_t* Bb = Bs[step & 1];
    #pragma unroll
    for (int j = 0; j < 4; ++j) {
      const bf16_t* bp = Bb + (j * 16 + l16) * BS_STRIDE + koff;
      v8bf b0 = *(const v8bf*)bp;
      v8bf b1 = *(const v8bf*)(bp + 16);
      v16bf bfrag;
      #pragma unroll
      for (int i = 0; i < 8; ++i) { bfrag[i] = b0[i]; bfrag[8 + i] = b1[i]; }
      acc[j] = __builtin_amdgcn_wmma_f32_16x16x32_bf16(
          false, afrag, false, bfrag, (short)0, acc[j], false, false);
    }
#if USE_ASYNC_COPY
    __builtin_amdgcn_s_wait_asynccnt(0);
#endif
    __syncthreads();
  }

  // ---- epilogue: C/D layout -> out[n][f][oh][ow] + bias ----
  float bv[8];
  #pragma unroll
  for (int v = 0; v < 8; ++v) bv[v] = bias[(wave << 4) + v + (half << 3)];
  #pragma unroll
  for (int j = 0; j < 4; ++j) {
    const int pp   = j * 16 + l16;
    const int ow_o = pp & 31;
    const int oh_o = oh0 + (pp >> 5);
    #pragma unroll
    for (int v = 0; v < 8; ++v) {
      const int f = (wave << 4) + v + (half << 3);
      out[(((size_t)n_img * F_OUT + f) * HW + oh_o) * HW + ow_o] = acc[j][v] + bv[v];
    }
  }
}

// ---------------- fallback main (small workspace): bounds-checked gather ----------------
// MODE 1: A packed ws, B f32 NCHW + cvt ; MODE 0: A f32 gather + cvt
template <int MODE>
__global__ __launch_bounds__(256)
void conv_slow(const float* __restrict__ x, const float* __restrict__ w,
               const bf16_t* __restrict__ wp, const float* __restrict__ bias,
               float* __restrict__ out) {
  __shared__ __attribute__((aligned(16))) bf16_t Bsl[BUF_ELEMS];

  const int t    = threadIdx.x;
  const int wave = t >> 5;
  const int lane = t & 31;
  const int half = lane >> 4;
  const int l16  = lane & 15;
  const int bx    = blockIdx.x;
  const int n_img = bx >> 4;
  const int oh0   = (bx & 15) << 1;
  const int p_st  = t & 63;
  const int cg    = t >> 6;
  const int ow_st = p_st & 31;
  const int oh_st = oh0 + (p_st >> 5);

  v8f acc[4] = { v8f{}, v8f{}, v8f{}, v8f{} };
  const int mrow = (wave << 4) + l16;
  bf16_t* const dstp = &Bsl[p_st * BS_STRIDE + cg * 8];

  for (int pos = 0; pos < 9; ++pos) {
    const int kh = pos / 3, kw = pos - kh * 3;
    const int ih = oh_st + kh - 1;
    const int iw = ow_st + kw - 1;
    const bool inb = ((unsigned)ih < HW) && ((unsigned)iw < HW);

    for (int cc = 0; cc < 4; ++cc) {
      union { unsigned short us[8]; uint4 q; } pk;
      if (inb) {
        const float* srcp = x + ((size_t)n_img * C_IN + cc * 32 + cg * 8) * (HW * HW)
                            + ih * HW + iw;
        #pragma unroll
        for (int i = 0; i < 8; ++i) pk.us[i] = f2bfbits(srcp[(size_t)i * (HW * HW)]);
      } else {
        pk.q = uint4{0u, 0u, 0u, 0u};
      }
      *(uint4*)dstp = pk.q;
      __syncthreads();

      v16bf afrag;
      if constexpr (MODE >= 1) {
        const bf16_t* ap = wp + ((size_t)((pos * 4 + cc) * 128 + mrow) * 32 + (half << 4));
        v8bf a0 = *(const v8bf*)ap;
        v8bf a1 = *(const v8bf*)(ap + 8);
        #pragma unroll
        for (int i = 0; i < 8; ++i) { afrag[i] = a0[i]; afrag[8 + i] = a1[i]; }
      } else {
        const float* wr = w + (size_t)mrow * K_TOT + pos;
        #pragma unroll
        for (int i = 0; i < 8; ++i) {
          afrag[i]     = f2bf(wr[(cc * 32 + (half << 3) + i) * 9]);
          afrag[8 + i] = f2bf(wr[(cc * 32 + 16 + (half << 3) + i) * 9]);
        }
      }

      const int koff = half << 3;
      #pragma unroll
      for (int j = 0; j < 4; ++j) {
        const bf16_t* bp = &Bsl[(j * 16 + l16) * BS_STRIDE + koff];
        v8bf b0 = *(const v8bf*)bp;
        v8bf b1 = *(const v8bf*)(bp + 16);
        v16bf bfrag;
        #pragma unroll
        for (int i = 0; i < 8; ++i) { bfrag[i] = b0[i]; bfrag[8 + i] = b1[i]; }
        acc[j] = __builtin_amdgcn_wmma_f32_16x16x32_bf16(
            false, afrag, false, bfrag, (short)0, acc[j], false, false);
      }
      __syncthreads();
    }
  }

  float bv[8];
  #pragma unroll
  for (int v = 0; v < 8; ++v) bv[v] = bias[(wave << 4) + v + (half << 3)];
  #pragma unroll
  for (int j = 0; j < 4; ++j) {
    const int pp   = j * 16 + l16;
    const int ow_o = pp & 31;
    const int oh_o = oh0 + (pp >> 5);
    #pragma unroll
    for (int v = 0; v < 8; ++v) {
      const int f = (wave << 4) + v + (half << 3);
      out[(((size_t)n_img * F_OUT + f) * HW + oh_o) * HW + ow_o] = acc[j][v] + bv[v];
    }
  }
}

extern "C" void kernel_launch(void* const* d_in, const int* in_sizes, int n_in,
                              void* d_out, int out_size, void* d_ws, size_t ws_size,
                              hipStream_t stream) {
  const float* x = (const float*)d_in[0];   // [32,128,32,32]
  const float* w = (const float*)d_in[1];   // [128,128,3,3]
  const float* b = (const float*)d_in[2];   // [128]
  float* out = (float*)d_out;               // [32,128,32,32]
  (void)in_sizes; (void)n_in; (void)out_size;

  bf16_t* wp = (bf16_t*)d_ws;
  bf16_t* xp = (bf16_t*)((char*)d_ws + WPACK_BYTES);

  const dim3 grid(32 * 16), block(256);

  if (ws_size >= (size_t)(WPACK_BYTES + XP_BYTES)) {
    pack_w_kernel<<<WPACK_ELEMS / 256, 256, 0, stream>>>(w, wp);
    nhwc_pad_kernel<<<32 * PHW, 256, 0, stream>>>(x, xp);
    conv_fast<<<grid, block, 0, stream>>>(xp, wp, b, out);
  } else if (ws_size >= (size_t)WPACK_BYTES) {
    pack_w_kernel<<<WPACK_ELEMS / 256, 256, 0, stream>>>(w, wp);
    conv_slow<1><<<grid, block, 0, stream>>>(x, w, wp, b, out);
  } else {
    conv_slow<0><<<grid, block, 0, stream>>>(x, w, nullptr, b, out);
  }
}
